// SimpleModel_29059748725144
// MI455X (gfx1250) — matrix operand order repaired and checked by reading, not compiled, not run on hardware
//
#include <hip/hip_runtime.h>

typedef __attribute__((ext_vector_type(16))) _Float16 v16h;
typedef __attribute__((ext_vector_type(8)))  float    v8f;

#define NN 50000
#define NE 600000
#define D  128

// ---------------------------------------------------------------------------
// Dense GEMM: out[n,o] = sum_k X[n,k] * W[o,k]  (+ bias, optional relu)
// WMMA f32<=f16 16x16x32. W (128x128) staged transposed as f16 in LDS.
// Block = 128 threads = 4 waves; each wave owns a 16-row x 128-col stripe.
// ---------------------------------------------------------------------------
__global__ __launch_bounds__(128) void gemm_wmma_kernel(
    const float* __restrict__ X, const float* __restrict__ W,
    const float* __restrict__ bias, float* __restrict__ out,
    int nrows, int do_relu)
{
  __shared__ _Float16 Wt[D * D]; // Wt[k][o] = W[o][k], 32 KB
  const int tid = threadIdx.x;
  for (int idx = tid; idx < D * D; idx += 128) {
    const int k = idx >> 7, o = idx & 127;
    Wt[idx] = (_Float16)W[o * D + k];
  }
  __syncthreads();

  const int wave = tid >> 5, lane = tid & 31;
  const int m = lane & 15, half = lane >> 4;
  const int rowBase = blockIdx.x * 64 + wave * 16;
  const int row = rowBase + m;
  const int rowc = row < nrows ? row : (nrows - 1);
  const float* xp = X + (long)rowc * D;

  v8f acc[8];
#pragma unroll
  for (int c = 0; c < 8; ++c) acc[c] = (v8f){0.f,0.f,0.f,0.f,0.f,0.f,0.f,0.f};

#pragma unroll
  for (int k0 = 0; k0 < D; k0 += 32) {
    // A tile (16x32 f16): lane (m,half): elems 0..7 -> K=half*8+t, 8..15 -> K=16+half*8+t
    v16h a;
#pragma unroll
    for (int t = 0; t < 8; ++t) {
      a[t]     = (_Float16)xp[k0 + half * 8 + t];
      a[t + 8] = (_Float16)xp[k0 + 16 + half * 8 + t];
    }
#pragma unroll
    for (int c = 0; c < 8; ++c) {
      // B tile (32x16 f16): lane (n,half): N = c*16+n, elems 0..7 -> K=half*8+t, 8..15 -> K=16+half*8+t
      v16h b;
      const _Float16* wp = &Wt[k0 * D + c * 16 + (lane & 15)];
#pragma unroll
      for (int t = 0; t < 8; ++t) {
        b[t]     = wp[(half * 8 + t) * D];
        b[t + 8] = wp[(16 + half * 8 + t) * D];
      }
      acc[c] = __builtin_amdgcn_wmma_f32_16x16x32_f16(
          false, a, false, b, (short)0, acc[c], false, false);
    }
  }

  // C/D layout: VGPR r, lane l -> M = r + 8*(l>>4), N = l&15
  const int n = lane & 15;
#pragma unroll
  for (int c = 0; c < 8; ++c) {
    const int o = c * 16 + n;
    const float bv = bias ? bias[o] : 0.0f;
#pragma unroll
    for (int r = 0; r < 8; ++r) {
      const int orow = rowBase + r + 8 * half;
      if (orow < nrows) {
        float v = acc[c][r] + bv;
        if (do_relu) v = v > 0.0f ? v : 0.0f;
        out[(long)orow * D + o] = v;
      }
    }
  }
}

// ---------------------------------------------------------------------------
// Utility kernels
// ---------------------------------------------------------------------------
__global__ void zero_kernel(float* p, long n) {
  long i = (long)blockIdx.x * blockDim.x + threadIdx.x;
  const long stride = (long)gridDim.x * blockDim.x;
  for (; i < n; i += stride) p[i] = 0.0f;
}

__global__ void init_scalars_kernel(unsigned long long* key, int* j, float* snuc) {
  *key = 0ull; *j = 0; *snuc = 1.0f;
}

__global__ void deg_kernel(const int* __restrict__ dst, float* __restrict__ deg) {
  int e = blockIdx.x * 256 + threadIdx.x;
  if (e < NE) atomicAdd(&deg[dst[e]], 1.0f);
}

__global__ void dinv_kernel(const float* __restrict__ deg, float* __restrict__ dinv) {
  int nidx = blockIdx.x * 256 + threadIdx.x;
  if (nidx < NN) {
    float d = deg[nidx];
    dinv[nidx] = d > 0.0f ? 1.0f / sqrtf(d) : 0.0f;
  }
}

// one wave (32 lanes) per edge; each lane moves 4 floats
__global__ void scatter_kernel(const float* __restrict__ msg, const int* __restrict__ src,
                               const int* __restrict__ dst, const float* __restrict__ dinv,
                               float* __restrict__ out)
{
  long gid = (long)blockIdx.x * blockDim.x + threadIdx.x;
  long e = gid >> 5;
  int lane = (int)(gid & 31);
  if (e >= NE) return;
  int s = src[e], d = dst[e];
  float nrm = dinv[s] * dinv[d];
  float4 v = *((const float4*)(msg + (long)s * D) + lane);
  float* op = out + (long)d * D + lane * 4;
  atomicAdd(op + 0, v.x * nrm);
  atomicAdd(op + 1, v.y * nrm);
  atomicAdd(op + 2, v.z * nrm);
  atomicAdd(op + 3, v.w * nrm);
}

__global__ void bias_relu_kernel(const float* __restrict__ in, const float* __restrict__ bias,
                                 float* __restrict__ out, int do_relu)
{
  long i = (long)blockIdx.x * 256 + threadIdx.x;
  if (i < (long)NN * D) {
    float v = in[i] + bias[(int)(i & 127)];
    if (do_relu) v = fmaxf(v, 0.0f);
    out[i] = v;
  }
}

// ---------------------------------------------------------------------------
// rank_diff building blocks
// ---------------------------------------------------------------------------
// G += H^T H (128x128), partial per block in registers, atomic add at end.
__global__ __launch_bounds__(256) void gram_kernel(const float* __restrict__ H,
                                                   float* __restrict__ G)
{
  __shared__ float rowbuf[D];
  const int tid = threadIdx.x;
  const int a = tid >> 1;
  const int b0 = (tid & 1) * 64;
  float acc[64];
#pragma unroll
  for (int t = 0; t < 64; ++t) acc[t] = 0.0f;
  const int rpb = (NN + gridDim.x - 1) / gridDim.x;
  const int r0 = blockIdx.x * rpb;
  const int r1 = (r0 + rpb) < NN ? (r0 + rpb) : NN;
  for (int r = r0; r < r1; ++r) {
    __syncthreads();
    if (tid < D) rowbuf[tid] = H[(long)r * D + tid];
    __syncthreads();
    const float av = rowbuf[a];
#pragma unroll
    for (int t = 0; t < 64; ++t) acc[t] += av * rowbuf[b0 + t];
  }
  for (int t = 0; t < 64; ++t) atomicAdd(&G[a * D + b0 + t], acc[t]);
}

// packed-key argmax over per-row |.| sums; ties -> smallest index (first max)
__global__ void row_argmax_kernel(const float* __restrict__ H, unsigned long long* key) {
  int nidx = blockIdx.x * 256 + threadIdx.x;
  if (nidx >= NN) return;
  const float* p = H + (long)nidx * D;
  float s = 0.0f;
  for (int k = 0; k < D; k += 4)
    s += fabsf(p[k]) + fabsf(p[k+1]) + fabsf(p[k+2]) + fabsf(p[k+3]);
  unsigned long long kk =
      ((unsigned long long)__float_as_uint(s) << 32) | (unsigned)(~(unsigned)nidx);
  atomicMax(key, kk);
}

__global__ __launch_bounds__(256) void colsum_kernel(const float* __restrict__ H,
                                                     float* __restrict__ csum)
{
  const int rpb = (NN + gridDim.x - 1) / gridDim.x;
  long start = (long)blockIdx.x * rpb * D;
  long endr = (long)(blockIdx.x + 1) * rpb;
  long end = (endr < NN ? endr : NN) * (long)D;
  float s = 0.0f;
  const int col = threadIdx.x & 127; // constant per-thread since stride 256, start%128==0
  for (long i = start + threadIdx.x; i < end; i += 256) s += fabsf(H[i]);
  atomicAdd(&csum[col], s);
}

__global__ void col_argmax_kernel(const float* __restrict__ csum, int* jout) {
  if (threadIdx.x == 0) {
    float best = csum[0]; int bi = 0;
    for (int k = 1; k < D; ++k) if (csum[k] > best) { best = csum[k]; bi = k; }
    *jout = bi;
  }
}

// G'' = G/s^2 - (v m0^T + m0 v^T)/(|m1||m0|) + m0 m0^T/|m0|^2
__global__ __launch_bounds__(128) void build_g2_kernel(
    const float* __restrict__ H, const float* __restrict__ G,
    const unsigned long long* key, const int* jptr,
    const float* snuc, float* __restrict__ G2)
{
  __shared__ float m0[D], v[D], red[D];
  const int tid = threadIdx.x;
  const int i = (int)(~(unsigned)(*key & 0xFFFFFFFFull));
  const int j = *jptr;
  const float s = *snuc;
  const float invs2 = 1.0f / (s * s);
  const float sgn = (H[(long)i * D + j] < 0.0f) ? -1.0f : 1.0f;
  m0[tid] = sgn * H[(long)i * D + tid] / s;
  v[tid]  = G[tid * D + j] * invs2;
  red[tid] = m0[tid] * m0[tid];
  __syncthreads();
  for (int off = 64; off > 0; off >>= 1) {
    if (tid < off) red[tid] += red[tid + off];
    __syncthreads();
  }
  const float m0n2 = red[0];
  const float m1n2 = G[j * D + j] * invs2;
  const float inv_m1m0 = 1.0f / sqrtf(m1n2 * m0n2);
  const float inv_m0n2 = 1.0f / m0n2;
  const int a = tid;
  for (int b = 0; b < D; ++b) {
    G2[a * D + b] = G[a * D + b] * invs2
                  - (v[a] * m0[b] + m0[a] * v[b]) * inv_m1m0
                  + m0[a] * m0[b] * inv_m0n2;
  }
}

// nuclear norm via parallel cyclic Jacobi on 128x128 symmetric matrix in LDS.
// 64 disjoint rotations/round (round-robin pairing), row phase then col phase.
__global__ __launch_bounds__(256) void nuc_kernel(const float* __restrict__ Gin,
                                                  float* __restrict__ out)
{
  extern __shared__ float smem[];
  float* Amat = smem;            // 128*129 (padded rows)
  float* cs   = smem + D * 129;  // 64
  float* sn   = cs + 64;         // 64
  float* red  = sn + 64;         // 128
  int*   pp   = (int*)(red + 128); // 64
  int*   qq   = pp + 64;           // 64
  const int tid = threadIdx.x;
#define AM(r, c) Amat[(r) * 129 + (c)]
  for (int idx = tid; idx < D * D; idx += 256) AM(idx >> 7, idx & 127) = Gin[idx];
  __syncthreads();
  for (int sweep = 0; sweep < 12; ++sweep) {
    for (int round = 0; round < 127; ++round) {
      if (tid < 64) {
        int p, q;
        if (tid == 0) { p = 127; q = round; }
        else { p = (round + tid) % 127; q = (round + 127 - tid) % 127; }
        const float app = AM(p, p), aqq = AM(q, q), apq = AM(p, q);
        float c, s;
        if (fabsf(apq) < 1e-36f) { c = 1.0f; s = 0.0f; }
        else {
          float th = (aqq - app) / (2.0f * apq);
          float t = 1.0f / (fabsf(th) + sqrtf(1.0f + th * th));
          t = th < 0.0f ? -t : t;
          c = 1.0f / sqrtf(1.0f + t * t);
          s = t * c;
        }
        pp[tid] = p; qq[tid] = q; cs[tid] = c; sn[tid] = s;
      }
      __syncthreads();
      for (int item = tid; item < 64 * D; item += 256) { // A <- J^T A
        const int pr = item >> 7, col = item & 127;
        const int p = pp[pr], q = qq[pr];
        const float c = cs[pr], s = sn[pr];
        const float ap = AM(p, col), aq = AM(q, col);
        AM(p, col) = c * ap - s * aq;
        AM(q, col) = s * ap + c * aq;
      }
      __syncthreads();
      for (int item = tid; item < 64 * D; item += 256) { // A <- A J
        const int pr = item >> 7, rw = item & 127;
        const int p = pp[pr], q = qq[pr];
        const float c = cs[pr], s = sn[pr];
        const float ap = AM(rw, p), aq = AM(rw, q);
        AM(rw, p) = c * ap - s * aq;
        AM(rw, q) = s * ap + c * aq;
      }
      __syncthreads();
    }
  }
  if (tid < D) red[tid] = sqrtf(fmaxf(AM(tid, tid), 0.0f));
  __syncthreads();
  for (int off = 64; off > 0; off >>= 1) {
    if (tid < off) red[tid] += red[tid + off];
    __syncthreads();
  }
  if (tid == 0) *out = red[0];
#undef AM
}

// ---------------------------------------------------------------------------
extern "C" void kernel_launch(void* const* d_in, const int* in_sizes, int n_in,
                              void* d_out, int out_size, void* d_ws, size_t ws_size,
                              hipStream_t stream)
{
  (void)in_sizes; (void)n_in; (void)out_size; (void)ws_size;
  const float* x     = (const float*)d_in[0];
  const int*   eidx  = (const int*)d_in[1];
  const float* enc_W = (const float*)d_in[2];
  const float* enc_b = (const float*)d_in[3];
  const float* Wl[3] = {(const float*)d_in[4], (const float*)d_in[6], (const float*)d_in[8]};
  const float* bl[3] = {(const float*)d_in[5], (const float*)d_in[7], (const float*)d_in[9]};
  const int* src = eidx;
  const int* dst = eidx + NE;

  float* out   = (float*)d_out;
  float* stats = out + (long)NN * D;

  float* ws   = (float*)d_ws;
  float* bufA = ws;                         // 6,400,000
  float* bufB = ws + 6400000L;              // 6,400,000
  float* deg  = ws + 12800000L;             // 50,000
  float* dinv = ws + 12850000L;             // 50,000
  float* G    = ws + 12900096L;             // 16,384
  float* G2   = G + 16384;                  // 16,384
  float* csum = G2 + 16384;                 // 128
  float* snuc = csum + 128;                 // 1
  unsigned long long* key = (unsigned long long*)(csum + 136); // 8B-aligned
  int* jidx = (int*)(csum + 140);

  const size_t nuc_lds = (size_t)(D * 129 + 64 + 64 + 128) * 4 + 64 * 4 * 2; // 67584 B

  auto rank_diff = [&](const float* H, float* statdst) {
    zero_kernel<<<64, 256, 0, stream>>>(G, 16384L);
    zero_kernel<<<1, 128, 0, stream>>>(csum, 128L);
    init_scalars_kernel<<<1, 1, 0, stream>>>(key, jidx, snuc);
    gram_kernel<<<200, 256, 0, stream>>>(H, G);
    row_argmax_kernel<<<(NN + 255) / 256, 256, 0, stream>>>(H, key);
    colsum_kernel<<<200, 256, 0, stream>>>(H, csum);
    col_argmax_kernel<<<1, 32, 0, stream>>>(csum, jidx);
    nuc_kernel<<<1, 256, nuc_lds, stream>>>(G, snuc);
    build_g2_kernel<<<1, 128, 0, stream>>>(H, G, key, jidx, snuc, G2);
    nuc_kernel<<<1, 256, nuc_lds, stream>>>(G2, statdst);
  };

  // Encoder: h = x @ enc_W^T + enc_b
  gemm_wmma_kernel<<<(NN + 63) / 64, 128, 0, stream>>>(x, enc_W, enc_b, bufA, NN, 0);
  rank_diff(bufA, stats + 0);

  // Degrees (same for all layers)
  zero_kernel<<<200, 256, 0, stream>>>(deg, (long)NN);
  deg_kernel<<<(NE + 255) / 256, 256, 0, stream>>>(dst, deg);
  dinv_kernel<<<(NN + 255) / 256, 256, 0, stream>>>(deg, dinv);

  for (int l = 0; l < 3; ++l) {
    gemm_wmma_kernel<<<(NN + 63) / 64, 128, 0, stream>>>(bufA, Wl[l], nullptr, bufB, NN, 0);
    zero_kernel<<<2048, 256, 0, stream>>>(bufA, (long)NN * D);
    scatter_kernel<<<((long)NE * 32 + 255) / 256, 256, 0, stream>>>(bufB, src, dst, dinv, bufA);
    float* tgt = (l < 2) ? bufA : out;
    bias_relu_kernel<<<(int)(((long)NN * D + 255) / 256), 256, 0, stream>>>(
        bufA, bl[l], tgt, (l < 2) ? 1 : 0);
    rank_diff(tgt, stats + 1 + l);
  }
}
